// CausalSelfAttention_76304388981197
// MI455X (gfx1250) — matrix operand order repaired and checked by reading, not compiled, not run on hardware
//
#include <hip/hip_runtime.h>
#include <stdint.h>

// ---------------------------------------------------------------------------
// CDNA5 (gfx1250, wave32) causal self-attention, bf16 WMMA everywhere.
// Round 2: DPP 16-lane reductions (no ds_bpermute), async global->LDS K/V
// staging with double buffering (ASYNCcnt), V kept transposed in global.
// ---------------------------------------------------------------------------

typedef __bf16 bf16_t;
typedef __attribute__((ext_vector_type(16))) __bf16 v16bf;
typedef __attribute__((ext_vector_type(8)))  __bf16 v8bf;
typedef __attribute__((ext_vector_type(8)))  float  v8f;

#define D_MODEL 1024
#define SEQ     2048
#define BATCH   2
#define HEADS   16
#define DK      64
#define MROWS   (BATCH * SEQ)   // 4096

// D = A(16x32) * B(32x16) + C, bf16 in / f32 accumulate
static __device__ inline v8f wmma_bf16(v16bf a, v16bf b, v8f c) {
    return __builtin_amdgcn_wmma_f32_16x16x32_bf16(
        /*neg_a=*/false, a, /*neg_b=*/false, b,
        /*c_mod=*/(short)0, c, /*reuse_a=*/false, /*reuse_b=*/false);
}

// A fragment (16x32, row-major source): lane row = lane&15.
// lanes 0-15 hold K = {0..7, 16..23}; lanes 16-31 hold K = {8..15, 24..31}.
static __device__ inline v16bf frag_a(const bf16_t* rowPtr, int half) {
    v8bf lo = *reinterpret_cast<const v8bf*>(rowPtr + half * 8);
    v8bf hi = *reinterpret_cast<const v8bf*>(rowPtr + 16 + half * 8);
    v16bf r;
#pragma unroll
    for (int i = 0; i < 8; ++i) { r[i] = lo[i]; r[i + 8] = hi[i]; }
    return r;
}

// B fragment (32x16): lane col n = lane&15; colPtr -> 32-long K-run for col n.
// lanes 0-15 hold K = {0..7, 16..23}; lanes 16-31 hold K = {8..15, 24..31}.
static __device__ inline v16bf frag_b(const bf16_t* colPtr, int half) {
    v8bf lo = *reinterpret_cast<const v8bf*>(colPtr + half * 8);
    v8bf hi = *reinterpret_cast<const v8bf*>(colPtr + 16 + half * 8);
    v16bf r;
#pragma unroll
    for (int i = 0; i < 8; ++i) { r[i] = lo[i]; r[i + 8] = hi[i]; }
    return r;
}

// 16-lane butterfly reductions, pure VALU DPP (no LDS traffic / waits).
// quad_perm(1,0,3,2)=0xB1, quad_perm(2,3,0,1)=0x4E,
// row_half_mirror=0x141, row_mirror=0x140.
static __device__ inline float row16_max(float x) {
    x = fmaxf(x, __int_as_float(__builtin_amdgcn_update_dpp(
                     0, __float_as_int(x), 0xB1, 0xF, 0xF, true)));
    x = fmaxf(x, __int_as_float(__builtin_amdgcn_update_dpp(
                     0, __float_as_int(x), 0x4E, 0xF, 0xF, true)));
    x = fmaxf(x, __int_as_float(__builtin_amdgcn_update_dpp(
                     0, __float_as_int(x), 0x141, 0xF, 0xF, true)));
    x = fmaxf(x, __int_as_float(__builtin_amdgcn_update_dpp(
                     0, __float_as_int(x), 0x140, 0xF, 0xF, true)));
    return x;
}
static __device__ inline float row16_sum(float x) {
    x += __int_as_float(__builtin_amdgcn_update_dpp(
             0, __float_as_int(x), 0xB1, 0xF, 0xF, true));
    x += __int_as_float(__builtin_amdgcn_update_dpp(
             0, __float_as_int(x), 0x4E, 0xF, 0xF, true));
    x += __int_as_float(__builtin_amdgcn_update_dpp(
             0, __float_as_int(x), 0x141, 0xF, 0xF, true));
    x += __int_as_float(__builtin_amdgcn_update_dpp(
             0, __float_as_int(x), 0x140, 0xF, 0xF, true));
    return x;
}

// ---------------------------------------------------------------------------
// GEMM: out[i][j] = sum_k A[i][k] * W[j][k] + bias[j]   (y = A @ W^T + b)
// MODE 0: A = f32 x. z=0 -> Q [B,H,S,Dk] bf16; z=1 -> K [B,H,S,Dk] bf16;
//         z=2 -> V TRANSPOSED [B,H,Dk,S] bf16 (vectorized 16B stores).
// MODE 1: A = bf16 attn, writes f32 + bias into d_out [B*S, D].
// Block 256 threads = 8 waves; tile 128(M) x 64(N), K-step 32; wave 2x2 WMMA.
// ---------------------------------------------------------------------------
template <int MODE>
__global__ __launch_bounds__(256) void gemm_bf16_wmma(
    const float* __restrict__ Xf, const bf16_t* __restrict__ Xbf,
    const float* __restrict__ W0, const float* __restrict__ W1,
    const float* __restrict__ W2,
    const float* __restrict__ b0, const float* __restrict__ b1,
    const float* __restrict__ b2,
    bf16_t* __restrict__ O0, bf16_t* __restrict__ O1, bf16_t* __restrict__ O2,
    float* __restrict__ Fo)
{
    __shared__ alignas(16) bf16_t As[128 * 40];
    __shared__ alignas(16) bf16_t Bs[64 * 40];

    const int tid  = threadIdx.x;
    const int lane = tid & 31;
    const int wid  = tid >> 5;
    const int l15  = lane & 15;
    const int half = lane >> 4;
    const int wm   = wid >> 1;
    const int wn   = wid & 1;
    const int rowBase = blockIdx.y * 128;
    const int nBase   = blockIdx.x * 64;
    const int z       = blockIdx.z;

    const float* W    = (z == 0) ? W0 : (z == 1) ? W1 : W2;
    const float* bias = (z == 0) ? b0 : (z == 1) ? b1 : b2;
    bf16_t*      dst  = (z == 0) ? O0 : (z == 1) ? O1 : O2;

    const int a_r  = tid >> 1;
    const int a_kc = (tid & 1) * 16;
    const int b_r  = tid >> 2;
    const int b_kc = (tid & 3) * 8;

    v8f acc[2][2] = {};

    for (int k0 = 0; k0 < D_MODEL; k0 += 32) {
        __syncthreads();
        if (MODE == 0) {
            const float4* src = reinterpret_cast<const float4*>(
                Xf + (size_t)(rowBase + a_r) * D_MODEL + k0 + a_kc);
#pragma unroll
            for (int c = 0; c < 4; ++c) {
                float4 f = src[c];
                bf16_t* d = &As[a_r * 40 + a_kc + c * 4];
                d[0] = (bf16_t)f.x; d[1] = (bf16_t)f.y;
                d[2] = (bf16_t)f.z; d[3] = (bf16_t)f.w;
            }
            if (k0 + 32 < D_MODEL)
                __builtin_prefetch(Xf + (size_t)(rowBase + a_r) * D_MODEL +
                                   k0 + 32 + a_kc, 0, 1);
        } else {
            const v8bf* src = reinterpret_cast<const v8bf*>(
                Xbf + (size_t)(rowBase + a_r) * D_MODEL + k0 + a_kc);
            v8bf x0 = src[0], x1 = src[1];
            bf16_t* d = &As[a_r * 40 + a_kc];
#pragma unroll
            for (int j = 0; j < 8; ++j) { d[j] = x0[j]; d[8 + j] = x1[j]; }
            if (k0 + 32 < D_MODEL)
                __builtin_prefetch(Xbf + (size_t)(rowBase + a_r) * D_MODEL +
                                   k0 + 32 + a_kc, 0, 1);
        }
        {
            const float4* src = reinterpret_cast<const float4*>(
                W + (size_t)(nBase + b_r) * D_MODEL + k0 + b_kc);
            bf16_t* d = &Bs[b_r * 40 + b_kc];
#pragma unroll
            for (int c = 0; c < 2; ++c) {
                float4 f = src[c];
                d[c * 4 + 0] = (bf16_t)f.x; d[c * 4 + 1] = (bf16_t)f.y;
                d[c * 4 + 2] = (bf16_t)f.z; d[c * 4 + 3] = (bf16_t)f.w;
            }
            if (k0 + 32 < D_MODEL)
                __builtin_prefetch(W + (size_t)(nBase + b_r) * D_MODEL +
                                   k0 + 32 + b_kc, 0, 1);
        }
        __syncthreads();

        v16bf aF[2], bF[2];
#pragma unroll
        for (int mi = 0; mi < 2; ++mi)
            aF[mi] = frag_a(&As[(wm * 32 + mi * 16 + l15) * 40], half);
#pragma unroll
        for (int ni = 0; ni < 2; ++ni)
            bF[ni] = frag_b(&Bs[(wn * 32 + ni * 16 + l15) * 40], half);
#pragma unroll
        for (int mi = 0; mi < 2; ++mi)
#pragma unroll
            for (int ni = 0; ni < 2; ++ni)
                acc[mi][ni] = wmma_bf16(aF[mi], bF[ni], acc[mi][ni]);
    }

    // ---- epilogue ----
#pragma unroll
    for (int ni = 0; ni < 2; ++ni) {
        const int j  = nBase + wn * 32 + ni * 16 + l15;
        const float bj = bias[j];
        const int hh = j >> 6, dk = j & (DK - 1);
#pragma unroll
        for (int mi = 0; mi < 2; ++mi) {
            const int row0 = rowBase + wm * 32 + mi * 16 + 8 * half;
            if (MODE == 0 && z == 2) {
                // V transposed [B,H,Dk,S]: rows r are contiguous in S.
                const int bb = row0 >> 11, s0 = row0 & (SEQ - 1);
                v8bf pk;
#pragma unroll
                for (int r = 0; r < 8; ++r)
                    pk[r] = (bf16_t)(acc[mi][ni][r] + bj);
                *reinterpret_cast<v8bf*>(
                    &dst[(((size_t)bb * HEADS + hh) * DK + dk) * SEQ + s0]) = pk;
            } else {
#pragma unroll
                for (int r = 0; r < 8; ++r) {
                    const int row = row0 + r;
                    const float val = acc[mi][ni][r] + bj;
                    if (MODE == 0) {
                        const int bb = row >> 11, s = row & (SEQ - 1);
                        dst[((((size_t)bb * HEADS + hh) * SEQ) + s) * DK + dk] =
                            (bf16_t)val;
                    } else {
                        Fo[(size_t)row * D_MODEL + j] = val;
                    }
                }
            }
        }
    }
}

// ---------------------------------------------------------------------------
// Async stage of one key tile (32 keys): K [key][dk] and V^T [dk][key] tiles,
// via GLOBAL_LOAD_ASYNC_TO_LDS_B128 (GVS mode, tracked by ASYNCcnt).
// Each thread issues 2 K-chunks + 2 V-chunks (16B each) => 4 per wave.
// ---------------------------------------------------------------------------
static __device__ inline void async_stage_tile(uint64_t kBase, uint64_t vBase,
                                               int k0, int tid,
                                               uint32_t ksL, uint32_t vtL) {
    const int key = tid >> 2, dc = (tid & 3) * 16;           // K tile mapping
    const uint32_t gk = (uint32_t)(((k0 + key) * DK + dc) * 2);
    const uint32_t lk = ksL + (uint32_t)((key * 72 + dc) * 2);
    asm volatile("global_load_async_to_lds_b128 %0, %1, %2"
                 :: "v"(lk), "v"(gk), "s"(kBase) : "memory");
    asm volatile("global_load_async_to_lds_b128 %0, %1, %2"
                 :: "v"(lk + 16u), "v"(gk + 16u), "s"(kBase) : "memory");

    const int vdk = tid >> 1, kc = (tid & 1) * 16;           // V^T tile mapping
    const uint32_t gv = (uint32_t)((vdk * SEQ + k0 + kc) * 2);
    const uint32_t lv = vtL + (uint32_t)((vdk * 40 + kc) * 2);
    asm volatile("global_load_async_to_lds_b128 %0, %1, %2"
                 :: "v"(lv), "v"(gv), "s"(vBase) : "memory");
    asm volatile("global_load_async_to_lds_b128 %0, %1, %2"
                 :: "v"(lv + 16u), "v"(gv + 16u), "s"(vBase) : "memory");
}

// ---------------------------------------------------------------------------
// Causal flash attention. grid = (S/64, H, B), block = 128 (4 waves).
// Wave w owns query rows qt*64 + w*16 .. +15. Double-buffered async K/V tiles.
// V is stored transposed in global ([B,H,Dk,S]) so both stages are pure async.
// ---------------------------------------------------------------------------
__global__ __launch_bounds__(128) void attn_fwd_wmma(
    const bf16_t* __restrict__ Q, const bf16_t* __restrict__ K,
    const bf16_t* __restrict__ V, bf16_t* __restrict__ Ao)
{
    __shared__ alignas(16) bf16_t Ks[2 * 32 * 72];   // [buf][key][dk] pad 72
    __shared__ alignas(16) bf16_t Vt[2 * 64 * 40];   // [buf][dk][key] pad 40
    __shared__ alignas(16) bf16_t Ps[4 * 16 * 40];   // per-wave P

    const int tid  = threadIdx.x;
    const int lane = tid & 31;
    const int wid  = tid >> 5;
    const int l15  = lane & 15;
    const int half = lane >> 4;
    const int qt = blockIdx.x, h = blockIdx.y, b = blockIdx.z;
    const size_t bhS = ((size_t)b * HEADS + h) * SEQ;

    const uint64_t kBase = (uint64_t)(uintptr_t)(K + bhS * DK);
    const uint64_t vBase =
        (uint64_t)(uintptr_t)(V + ((size_t)b * HEADS + h) * (size_t)DK * SEQ);
    const uint32_t ksLds = (uint32_t)(uintptr_t)&Ks[0];
    const uint32_t vtLds = (uint32_t)(uintptr_t)&Vt[0];

    const int qw0 = qt * 64 + wid * 16;
    const bf16_t* qrow = Q + (bhS + qw0 + l15) * DK;
    const v16bf aQ0 = frag_a(qrow, half);
    const v16bf aQ1 = frag_a(qrow + 32, half);

    v8f o[4] = {};
    float rmax[8], rsum[8];
#pragma unroll
    for (int r = 0; r < 8; ++r) { rmax[r] = -3.0e38f; rsum[r] = 0.0f; }

    bf16_t* Pw = &Ps[wid * 16 * 40];
    const int nkt = 2 * qt + 2;            // keys 0 .. qt*64+63

    // prologue: stage tile 0 into buffer 0
    async_stage_tile(kBase, vBase, 0, tid, ksLds, vtLds);

    for (int kt = 0; kt < nkt; ++kt) {
        const int k0  = kt * 32;
        const int cur = kt & 1;
        __syncthreads();   // everyone done reading buf[cur^1] (iteration kt-1)
        if (kt + 1 < nkt) {
            async_stage_tile(kBase, vBase, k0 + 32, tid,
                             ksLds + (cur ^ 1) * (32 * 72 * 2),
                             vtLds + (cur ^ 1) * (64 * 40 * 2));
            asm volatile("s_wait_asynccnt 0x4" ::: "memory");  // tile kt landed
        } else {
            asm volatile("s_wait_asynccnt 0x0" ::: "memory");
        }
        __syncthreads();   // all waves' tile-kt async writes visible

        const bf16_t* KsC = &Ks[cur * 32 * 72];
        const bf16_t* VtC = &Vt[cur * 64 * 40];

        // ---- scores: S = Q(16x64) . K^T(64x32) -> two 16x16 subtiles ----
        v8f sc0 = {}, sc1 = {};
#pragma unroll
        for (int ks = 0; ks < 2; ++ks) {
            const v16bf aq = ks ? aQ1 : aQ0;
            const v16bf bk0 = frag_b(&KsC[l15 * 72 + ks * 32], half);
            const v16bf bk1 = frag_b(&KsC[(l15 + 16) * 72 + ks * 32], half);
            sc0 = wmma_bf16(aq, bk0, sc0);
            sc1 = wmma_bf16(aq, bk1, sc1);
        }

        // ---- online softmax (rows m = r + 8*half, col = lane&15) ----
#pragma unroll
        for (int r = 0; r < 8; ++r) {
            const int qi = qw0 + r + 8 * half;
            const int key0 = k0 + l15, key1 = key0 + 16;
            float s0 = sc0[r] * 0.125f;    // 1/sqrt(64)
            float s1 = sc1[r] * 0.125f;
            const bool m0 = key0 > qi, m1 = key1 > qi;
            if (m0) s0 = -3.0e38f;
            if (m1) s1 = -3.0e38f;
            const float mx   = row16_max(fmaxf(s0, s1));
            const float mnew = fmaxf(rmax[r], mx);
            const float corr = __expf(rmax[r] - mnew);
            const float p0 = m0 ? 0.0f : __expf(s0 - mnew);
            const float p1 = m1 ? 0.0f : __expf(s1 - mnew);
            const float ps = row16_sum(p0 + p1);
            rsum[r] = rsum[r] * corr + ps;
            rmax[r] = mnew;
#pragma unroll
            for (int d = 0; d < 4; ++d) o[d][r] *= corr;
            Pw[(r + 8 * half) * 40 + l15]      = (bf16_t)p0;
            Pw[(r + 8 * half) * 40 + l15 + 16] = (bf16_t)p1;
        }

        // ---- O += P(16x32) . V(32x64) ----
        const v16bf aP = frag_a(&Pw[l15 * 40], half);
#pragma unroll
        for (int d = 0; d < 4; ++d) {
            const v16bf bv = frag_b(&VtC[(d * 16 + l15) * 40], half);
            o[d] = wmma_bf16(aP, bv, o[d]);
        }
    }

    // ---- normalize and write [B, S, D] bf16 for the output projection ----
#pragma unroll
    for (int r = 0; r < 8; ++r) {
        const float inv = 1.0f / rsum[r];
        const int qi = qw0 + r + 8 * half;
        const size_t base = ((size_t)b * SEQ + qi) * D_MODEL + h * DK;
#pragma unroll
        for (int d = 0; d < 4; ++d)
            Ao[base + d * 16 + l15] = (bf16_t)(o[d][r] * inv);
    }
}

// ---------------------------------------------------------------------------
extern "C" void kernel_launch(void* const* d_in, const int* in_sizes, int n_in,
                              void* d_out, int out_size, void* d_ws,
                              size_t ws_size, hipStream_t stream) {
    (void)in_sizes; (void)n_in; (void)out_size; (void)ws_size;
    const float* x  = (const float*)d_in[0];
    const float* Wq = (const float*)d_in[1];
    const float* bq = (const float*)d_in[2];
    const float* Wk = (const float*)d_in[3];
    const float* bk = (const float*)d_in[4];
    const float* Wv = (const float*)d_in[5];
    const float* bv = (const float*)d_in[6];
    const float* Wo = (const float*)d_in[7];
    const float* bo = (const float*)d_in[8];
    float* out = (float*)d_out;

    char* ws = (char*)d_ws;
    const size_t bufBytes = (size_t)BATCH * HEADS * SEQ * DK * sizeof(bf16_t); // 8MB
    bf16_t* Qbf  = (bf16_t*)(ws + 0 * bufBytes);
    bf16_t* Kbf  = (bf16_t*)(ws + 1 * bufBytes);
    bf16_t* Vbf  = (bf16_t*)(ws + 2 * bufBytes);   // transposed [B,H,Dk,S]
    bf16_t* attn = (bf16_t*)(ws + 3 * bufBytes);

    gemm_bf16_wmma<0><<<dim3(D_MODEL / 64, MROWS / 128, 3), 256, 0, stream>>>(
        x, nullptr, Wq, Wk, Wv, bq, bk, bv, Qbf, Kbf, Vbf, nullptr);

    attn_fwd_wmma<<<dim3(SEQ / 64, HEADS, BATCH), 128, 0, stream>>>(
        Qbf, Kbf, Vbf, attn);

    gemm_bf16_wmma<1><<<dim3(D_MODEL / 64, MROWS / 128, 1), 256, 0, stream>>>(
        nullptr, attn, Wo, Wo, Wo, bo, bo, bo, nullptr, nullptr, nullptr, out);
}
